// DQN_GNN_2748779069596
// MI455X (gfx1250) — compile-verified, matrix operand order checked by reading, not run on hardware
//
#include <hip/hip_runtime.h>
#include <hip/hip_bf16.h>
#include <math.h>

#define NN      50000
#define FD      512
#define HD      512
#define NE      1600000
#define NI      10000
#define NP      10048      // NI padded to multiple of 64 for branch-free WMMA tiles
#define EPSV    1e-5f

typedef __attribute__((ext_vector_type(16))) __bf16 v16bf;
typedef __attribute__((ext_vector_type(8)))  __bf16 v8bf;
typedef __attribute__((ext_vector_type(8)))  float  v8f;

// ---------------------------------------------------------------- utilities
__global__ void k_fill_f32(float* __restrict__ p, float v, int n) {
    int i = blockIdx.x * blockDim.x + threadIdx.x;
    if (i < n) p[i] = v;
}

__global__ void k_fill_bf16(__bf16* __restrict__ p, int n) {
    int i = blockIdx.x * blockDim.x + threadIdx.x;
    if (i < n) p[i] = (__bf16)0.0f;
}

// dst[i] = i < n ? src[i] : 0   (zero-padded f32 copy)
__global__ void k_pad_vec(const float* __restrict__ src, float* __restrict__ dst,
                          int n, int np) {
    int i = blockIdx.x * blockDim.x + threadIdx.x;
    if (i < np) dst[i] = (i < n) ? src[i] : 0.0f;
}

__global__ void k_deg_count(const long long* __restrict__ dst, float* __restrict__ deg, int ne) {
    int e = blockIdx.x * blockDim.x + threadIdx.x;
    if (e < ne) atomicAdd(&deg[(int)dst[e]], 1.0f);
}

__global__ void k_dinv(const float* __restrict__ deg, float* __restrict__ dinv, int n) {
    int i = blockIdx.x * blockDim.x + threadIdx.x;
    if (i < n) dinv[i] = rsqrtf(deg[i]);   // deg >= 1 always (self loop)
}

__global__ void k_f32_to_bf16(const float* __restrict__ in, __bf16* __restrict__ out, long long n) {
    long long i = (long long)blockIdx.x * blockDim.x + threadIdx.x;
    long long stride = (long long)gridDim.x * blockDim.x;
    for (; i < n; i += stride) out[i] = (__bf16)in[i];
}

// Wt[n*K + k] = (bf16) W[k*N + n]   (K-major -> N-major for contiguous B frags)
__global__ void k_transpose_bf16(const float* __restrict__ W, __bf16* __restrict__ Wt, int K, int N) {
    long long t = (long long)blockIdx.x * blockDim.x + threadIdx.x;
    long long total = (long long)K * N;
    long long stride = (long long)gridDim.x * blockDim.x;
    for (; t < total; t += stride) {
        int k = (int)(t / N);
        int n = (int)(t % N);
        Wt[(size_t)n * K + k] = (__bf16)W[(size_t)k * N + n];
    }
}

// --------------------------------------------------------------- WMMA GEMM
// ISA 7.12.2 16-bit A 16x32 fragment: lane holds row (lane&15); half=lane>>4:
//   VGPR0-3 : K = half*8 + 0..7        VGPR4-7 : K = 16 + half*8 + 0..7
__device__ __forceinline__ v16bf load_frag16x32(const __bf16* __restrict__ p, int half) {
    const v8bf lo = *(const v8bf*)(p + half * 8);
    const v8bf hi = *(const v8bf*)(p + 16 + half * 8);
    return __builtin_shufflevector(lo, hi, 0, 1, 2, 3, 4, 5, 6, 7,
                                           8, 9, 10, 11, 12, 13, 14, 15);
}

// C[M,N] = A[M,K] * Bt[N,K]^T ; each wave computes a 16x64 tile (A reused x4).
// Requires M % 16 == 0, N % 64 == 0, K % 32 == 0. Branch-free WMMA path.
__global__ void __launch_bounds__(256)
k_wmma_gemm_bf16(const __bf16* __restrict__ A, const __bf16* __restrict__ Bt,
                 float* __restrict__ C, int M, int N, int K) {
    const int lane = threadIdx.x & 31;
    const int wave = (blockIdx.x * blockDim.x + threadIdx.x) >> 5;
    const int tilesN = N >> 6;
    const int tilesM = M >> 4;
    if (wave >= tilesM * tilesN) return;
    const int m0 = (wave / tilesN) << 4;
    const int n0 = (wave % tilesN) << 6;
    const int half = lane >> 4;
    const int lid  = lane & 15;

    v8f acc0 = {}, acc1 = {}, acc2 = {}, acc3 = {};
    const __bf16* arow = A  + (size_t)(m0 + lid) * K;
    const __bf16* b0   = Bt + (size_t)(n0 +  0 + lid) * K;
    const __bf16* b1   = Bt + (size_t)(n0 + 16 + lid) * K;
    const __bf16* b2   = Bt + (size_t)(n0 + 32 + lid) * K;
    const __bf16* b3   = Bt + (size_t)(n0 + 48 + lid) * K;
    for (int k0 = 0; k0 < K; k0 += 32) {
        v16bf af = load_frag16x32(arow + k0, half);
        acc0 = __builtin_amdgcn_wmma_f32_16x16x32_bf16(
            false, af, false, load_frag16x32(b0 + k0, half), (short)0, acc0, false, false);
        acc1 = __builtin_amdgcn_wmma_f32_16x16x32_bf16(
            false, af, false, load_frag16x32(b1 + k0, half), (short)0, acc1, false, false);
        acc2 = __builtin_amdgcn_wmma_f32_16x16x32_bf16(
            false, af, false, load_frag16x32(b2 + k0, half), (short)0, acc2, false, false);
        acc3 = __builtin_amdgcn_wmma_f32_16x16x32_bf16(
            false, af, false, load_frag16x32(b3 + k0, half), (short)0, acc3, false, false);
    }
    // C/D layout: lane has col n=(lane&15); VGPR r -> row m0 + r + 8*half
    float* crow = C + (size_t)(m0 + 8 * half) * N + n0 + lid;
#pragma unroll
    for (int r = 0; r < 8; ++r) {
        crow[(size_t)r * N +  0] = acc0[r];
        crow[(size_t)r * N + 16] = acc1[r];
        crow[(size_t)r * N + 32] = acc2[r];
        crow[(size_t)r * N + 48] = acc3[r];
    }
}

// ------------------------------------------------------- GCN message passing
__global__ void k_agg_init(const float* __restrict__ H, const float* __restrict__ dinv,
                           float* __restrict__ AGG) {
    long long t = (long long)blockIdx.x * blockDim.x + threadIdx.x;
    long long total = (long long)NN * HD;
    long long stride = (long long)gridDim.x * blockDim.x;
    for (; t < total; t += stride) {
        int node = (int)(t >> 9);               // /512
        float di = dinv[node];
        AGG[t] = H[t] * di * di;                // self-loop contribution
    }
}

__global__ void k_scatter(const long long* __restrict__ src, const long long* __restrict__ dst,
                          const float* __restrict__ dinv, const float* __restrict__ H,
                          float* __restrict__ AGG) {
    long long t = (long long)blockIdx.x * blockDim.x + threadIdx.x;
    long long total = (long long)NE * (HD / 4);
    long long stride = (long long)gridDim.x * blockDim.x;
    for (; t < total; t += stride) {
        int e = (int)(t >> 7);                  // HD/4 = 128 chunks per edge
        int f = (int)(t & 127) << 2;
        int s = (int)src[e], d = (int)dst[e];
        float nrm = dinv[s] * dinv[d];
        const float4 hv = *(const float4*)(H + (size_t)s * HD + f);
        float* out = AGG + (size_t)d * HD + f;
        atomicAdd(out + 0, hv.x * nrm);
        atomicAdd(out + 1, hv.y * nrm);
        atomicAdd(out + 2, hv.z * nrm);
        atomicAdd(out + 3, hv.w * nrm);
    }
}

// in-place (+bias) LayerNorm + activation; also emits bf16 A for next GEMM
// act: 0 = exact GELU, 1 = leaky_relu(0.01)
__global__ void k_ln_act(float* __restrict__ X, const float* __restrict__ bias,
                         const float* __restrict__ g, const float* __restrict__ b,
                         __bf16* __restrict__ Abf, int act) {
    const int lane = threadIdx.x & 31;
    const int node = blockIdx.x * (blockDim.x >> 5) + (threadIdx.x >> 5);
    if (node >= NN) return;
    float* row = X + (size_t)node * HD;
    float v[16];
    float s = 0.f;
#pragma unroll
    for (int j = 0; j < 16; ++j) {
        int f = lane + 32 * j;
        v[j] = row[f] + bias[f];
        s += v[j];
    }
#pragma unroll
    for (int off = 1; off < 32; off <<= 1) s += __shfl_xor(s, off, 32);
    float mu = s * (1.f / HD);
    float vs = 0.f;
#pragma unroll
    for (int j = 0; j < 16; ++j) { float d = v[j] - mu; vs += d * d; }
#pragma unroll
    for (int off = 1; off < 32; off <<= 1) vs += __shfl_xor(vs, off, 32);
    float rinv = rsqrtf(vs * (1.f / HD) + EPSV);
#pragma unroll
    for (int j = 0; j < 16; ++j) {
        int f = lane + 32 * j;
        float y = (v[j] - mu) * rinv * g[f] + b[f];
        float o = (act == 0) ? 0.5f * y * (1.f + erff(y * 0.70710678118f))
                             : (y < 0.f ? 0.01f * y : y);
        row[f] = o;
        Abf[(size_t)node * HD + f] = (__bf16)o;
    }
}

// -------------------------------------------------------------- item branch
__global__ void k_bn_stats(const float* __restrict__ item_emb,
                           float* __restrict__ mu, float* __restrict__ rinv) {
    __shared__ float ssum[256], ssq[256];
    int c = blockIdx.x;                          // one block per column
    float s = 0.f, q = 0.f;
    for (int i = threadIdx.x; i < NI; i += 256) {
        float v = item_emb[(size_t)i * HD + c];
        s += v; q += v * v;
    }
    ssum[threadIdx.x] = s; ssq[threadIdx.x] = q;
    __syncthreads();
    for (int o = 128; o > 0; o >>= 1) {
        if (threadIdx.x < o) { ssum[threadIdx.x] += ssum[threadIdx.x + o];
                               ssq[threadIdx.x]  += ssq[threadIdx.x + o]; }
        __syncthreads();
    }
    if (threadIdx.x == 0) {
        float m = ssum[0] * (1.f / NI);
        float var = ssq[0] * (1.f / NI) - m * m;
        mu[c] = m;
        rinv[c] = rsqrtf(var + EPSV);
    }
}

__global__ void k_build_combined(const float* __restrict__ item_emb,
                                 const float* __restrict__ mu, const float* __restrict__ rinv,
                                 const float* __restrict__ bng, const float* __restrict__ bnb,
                                 const float* __restrict__ rsu,     // X row 0
                                 __bf16* __restrict__ comb) {
    long long t = (long long)blockIdx.x * blockDim.x + threadIdx.x;
    long long total = (long long)NI * HD;
    if (t >= total) return;
    int i = (int)(t >> 9), c = (int)(t & (HD - 1));
    float v = (item_emb[t] - mu[c]) * rinv[c] * bng[c] + bnb[c];
    comb[(size_t)i * (2 * HD) + c]      = (__bf16)v;
    comb[(size_t)i * (2 * HD) + HD + c] = (__bf16)rsu[c];
}

__global__ void k_copy_rsu(const float* __restrict__ X, float* __restrict__ out) {
    int c = blockIdx.x * blockDim.x + threadIdx.x;
    if (c < HD) out[c] = X[c];
}

__global__ void k_qinit(float* __restrict__ q, const float* __restrict__ b2) {
    int i = blockIdx.x * blockDim.x + threadIdx.x;
    if (i < NI) q[i] = b2[0];
}

// fused: q[m] += sum_n relu( (comb @ W1)[m,n] + b1[n] ) * W2[n]
// N (padded) % 64 == 0; pad region of Bt/b1/W2 is zero so it contributes 0.
__global__ void __launch_bounds__(256)
k_wmma_mlp_fused(const __bf16* __restrict__ A, const __bf16* __restrict__ Bt,
                 const float* __restrict__ b1v, const float* __restrict__ w2v,
                 float* __restrict__ q, int M, int N, int K) {
    const int lane = threadIdx.x & 31;
    const int wave = (blockIdx.x * blockDim.x + threadIdx.x) >> 5;
    const int tilesN = N >> 6;
    const int tilesM = M >> 4;
    if (wave >= tilesM * tilesN) return;
    const int m0 = (wave / tilesN) << 4;
    const int n0 = (wave % tilesN) << 6;
    const int half = lane >> 4;
    const int lid  = lane & 15;

    v8f acc0 = {}, acc1 = {}, acc2 = {}, acc3 = {};
    const __bf16* arow = A  + (size_t)(m0 + lid) * K;
    const __bf16* b0   = Bt + (size_t)(n0 +  0 + lid) * K;
    const __bf16* b1p  = Bt + (size_t)(n0 + 16 + lid) * K;
    const __bf16* b2p  = Bt + (size_t)(n0 + 32 + lid) * K;
    const __bf16* b3p  = Bt + (size_t)(n0 + 48 + lid) * K;
    for (int k0 = 0; k0 < K; k0 += 32) {
        v16bf af = load_frag16x32(arow + k0, half);
        acc0 = __builtin_amdgcn_wmma_f32_16x16x32_bf16(
            false, af, false, load_frag16x32(b0  + k0, half), (short)0, acc0, false, false);
        acc1 = __builtin_amdgcn_wmma_f32_16x16x32_bf16(
            false, af, false, load_frag16x32(b1p + k0, half), (short)0, acc1, false, false);
        acc2 = __builtin_amdgcn_wmma_f32_16x16x32_bf16(
            false, af, false, load_frag16x32(b2p + k0, half), (short)0, acc2, false, false);
        acc3 = __builtin_amdgcn_wmma_f32_16x16x32_bf16(
            false, af, false, load_frag16x32(b3p + k0, half), (short)0, acc3, false, false);
    }
    // epilogue: relu(+bias) * W2, summed over the 64 columns of this tile
    float part[8];
    {
        const float bb0 = b1v[n0 + lid],      ww0 = w2v[n0 + lid];
        const float bb1 = b1v[n0 + 16 + lid], ww1 = w2v[n0 + 16 + lid];
        const float bb2 = b1v[n0 + 32 + lid], ww2 = w2v[n0 + 32 + lid];
        const float bb3 = b1v[n0 + 48 + lid], ww3 = w2v[n0 + 48 + lid];
#pragma unroll
        for (int r = 0; r < 8; ++r) {
            float h0 = acc0[r] + bb0, h1 = acc1[r] + bb1;
            float h2 = acc2[r] + bb2, h3 = acc3[r] + bb3;
            part[r] = (h0 > 0.f ? h0 : 0.f) * ww0 + (h1 > 0.f ? h1 : 0.f) * ww1 +
                      (h2 > 0.f ? h2 : 0.f) * ww2 + (h3 > 0.f ? h3 : 0.f) * ww3;
        }
    }
#pragma unroll
    for (int r = 0; r < 8; ++r) {
        float v = part[r];
        v += __shfl_xor(v, 1, 32);
        v += __shfl_xor(v, 2, 32);
        v += __shfl_xor(v, 4, 32);
        v += __shfl_xor(v, 8, 32);     // reduce across the 16 lanes sharing row set
        if (lid == 0) atomicAdd(&q[m0 + r + 8 * half], v);
    }
}

// ------------------------------------------------------------------- launch
extern "C" void kernel_launch(void* const* d_in, const int* in_sizes, int n_in,
                              void* d_out, int out_size, void* d_ws, size_t ws_size,
                              hipStream_t stream) {
    (void)in_sizes; (void)n_in; (void)out_size; (void)ws_size;
    const float*     node_feature = (const float*)d_in[0];
    const long long* edge_index   = (const long long*)d_in[1];
    const float* W[4]   = {(const float*)d_in[3],  (const float*)d_in[7],
                           (const float*)d_in[11], (const float*)d_in[15]};
    const float* bia[4] = {(const float*)d_in[4],  (const float*)d_in[8],
                           (const float*)d_in[12], (const float*)d_in[16]};
    const float* lng[4] = {(const float*)d_in[5],  (const float*)d_in[9],
                           (const float*)d_in[13], (const float*)d_in[17]};
    const float* lnb[4] = {(const float*)d_in[6],  (const float*)d_in[10],
                           (const float*)d_in[14], (const float*)d_in[18]};
    const float* item_emb = (const float*)d_in[19];
    const float* bn_g   = (const float*)d_in[20];
    const float* bn_b   = (const float*)d_in[21];
    const float* mlp_W1 = (const float*)d_in[22];
    const float* mlp_b1 = (const float*)d_in[23];
    const float* mlp_W2 = (const float*)d_in[24];
    const float* mlp_b2 = (const float*)d_in[25];
    float* out = (float*)d_out;                  // [0..9999]=q, [10000..10511]=rsu

    // -------- workspace carve (≈ 300 MB)
    char* ws = (char*)d_ws;
    auto carve = [&](size_t bytes) -> void* {
        void* p = (void*)ws;
        ws += (bytes + 255) & ~(size_t)255;
        return p;
    };
    float*  deg   = (float*) carve((size_t)NN * 4);
    float*  dinv  = (float*) carve((size_t)NN * 4);
    __bf16* Abf   = (__bf16*)carve((size_t)NN * HD * 2);
    __bf16* Wt    = (__bf16*)carve((size_t)HD * HD * 2);
    float*  H     = (float*) carve((size_t)NN * HD * 4);
    float*  AGG   = (float*) carve((size_t)NN * HD * 4);
    __bf16* comb  = (__bf16*)carve((size_t)NI * 2 * HD * 2);
    __bf16* W1t   = (__bf16*)carve((size_t)NP * 2 * HD * 2);   // padded rows
    float*  b1pad = (float*) carve((size_t)NP * 4);
    float*  w2pad = (float*) carve((size_t)NP * 4);
    float*  bnmu  = (float*) carve((size_t)HD * 4);
    float*  bnri  = (float*) carve((size_t)HD * 4);

    const long long src_off = 0, dst_off = NE;

    // -------- GCN normalization: deg = 1 + indeg ; dinv = rsqrt(deg)
    k_fill_f32<<<(NN + 255) / 256, 256, 0, stream>>>(deg, 1.0f, NN);
    k_deg_count<<<(NE + 255) / 256, 256, 0, stream>>>(edge_index + dst_off, deg, NE);
    k_dinv<<<(NN + 255) / 256, 256, 0, stream>>>(deg, dinv, NN);

    // -------- first bf16 activation matrix
    const long long NF = (long long)NN * FD;
    k_f32_to_bf16<<<(unsigned)((NF + 255) / 256), 256, 0, stream>>>(node_feature, Abf, NF);

    // -------- 4 GCN layers
    const int gemmWaves  = (NN / 16) * (HD / 64);            // 25000
    const int gemmBlocks = (gemmWaves * 32 + 255) / 256;     // 3125
    const long long scat = (long long)NE * (HD / 4);
    for (int l = 0; l < 4; ++l) {
        k_transpose_bf16<<<(HD * HD + 255) / 256, 256, 0, stream>>>(W[l], Wt, HD, HD);
        k_wmma_gemm_bf16<<<gemmBlocks, 256, 0, stream>>>(Abf, Wt, H, NN, HD, HD);
        k_agg_init<<<(unsigned)((NF + 255) / 256), 256, 0, stream>>>(H, dinv, AGG);
        k_scatter<<<(unsigned)((scat + 255) / 256), 256, 0, stream>>>(
            edge_index + src_off, edge_index + dst_off, dinv, H, AGG);
        k_ln_act<<<(NN + 7) / 8, 256, 0, stream>>>(AGG, bia[l], lng[l], lnb[l], Abf,
                                                   l < 3 ? 0 : 1);
    }

    // -------- rsu embedding output
    k_copy_rsu<<<2, 256, 0, stream>>>(AGG, out + NI);

    // -------- item BatchNorm + combined [items_bn || rsu] in bf16
    k_bn_stats<<<HD, 256, 0, stream>>>(item_emb, bnmu, bnri);
    const long long icnt = (long long)NI * HD;
    k_build_combined<<<(unsigned)((icnt + 255) / 256), 256, 0, stream>>>(
        item_emb, bnmu, bnri, bn_g, bn_b, AGG, comb);

    // -------- MLP: transpose W1 (pad rows zeroed), pad b1/W2, init q with b2,
    //          fused WMMA GEMM + relu + dot(W2)
    const int padElems = (NP - NI) * 2 * HD;                 // 48 * 1024
    k_fill_bf16<<<(padElems + 255) / 256, 256, 0, stream>>>(
        W1t + (size_t)NI * 2 * HD, padElems);
    const long long w1cnt = (long long)(2 * HD) * NI;
    k_transpose_bf16<<<(unsigned)((w1cnt + 255) / 256), 256, 0, stream>>>(
        mlp_W1, W1t, 2 * HD, NI);
    k_pad_vec<<<(NP + 255) / 256, 256, 0, stream>>>(mlp_b1, b1pad, NI, NP);
    k_pad_vec<<<(NP + 255) / 256, 256, 0, stream>>>(mlp_W2, w2pad, NI, NP);
    k_qinit<<<(NI + 255) / 256, 256, 0, stream>>>(out, mlp_b2);
    const int mlpWaves  = (NI / 16) * (NP / 64);             // 625 * 157
    const int mlpBlocks = (mlpWaves * 32 + 255) / 256;
    k_wmma_mlp_fused<<<mlpBlocks, 256, 0, stream>>>(comb, W1t, b1pad, w2pad, out,
                                                    NI, NP, 2 * HD);
}
